// DeformConv3DBlock_49100066128566
// MI455X (gfx1250) — compile-verified, matrix operand order checked by reading, not run on hardware
//
#include <hip/hip_runtime.h>
#include <hip/hip_bf16.h>

// ---- problem constants ----
constexpr int B_    = 2;
constexpr int CIN_  = 64;
constexpr int COUT_ = 96;
constexpr int T_    = 8;
constexpr int H_    = 48;
constexpr int W_    = 48;
constexpr int NPOS_ = T_ * H_ * W_;      // 18432
constexpr int K27_  = 27;
constexpr int KDIM_ = K27_ * CIN_;       // 1728
constexpr int KBLK_ = KDIM_ / 32;        // 54 k-blocks of 32
constexpr int NT_   = 32;                // positions per block (2 WMMA sub-tiles)
constexpr int NTILES_ = NPOS_ / NT_;     // 576
constexpr int LDS_WORDS_ = 2 * KBLK_ * 256;            // two B sub-tiles
constexpr unsigned LDS_BYTES_ = LDS_WORDS_ * 4;        // 110592 B dynamic LDS

typedef __attribute__((ext_vector_type(16))) __bf16 v16bf;
typedef __attribute__((ext_vector_type(8)))  float  v8f;

__device__ __forceinline__ unsigned short f2bf(float f) {
  unsigned u = __float_as_uint(f);
  u += 0x7fffu + ((u >> 16) & 1u);        // round-to-nearest-even
  return (unsigned short)(u >> 16);
}

// ------------------------------------------------------------------
// Stage 0a: x [B][C][T][H][W] f32  ->  xh [B][n][C] bf16 (channels-last)
// ------------------------------------------------------------------
__global__ __launch_bounds__(256) void k_convert_x(const float* __restrict__ x,
                                                   unsigned short* __restrict__ xh) {
  int e = blockIdx.x * 256 + threadIdx.x;
  if (e >= B_ * CIN_ * NPOS_) return;
  int n = e % NPOS_;
  int c = (e / NPOS_) % CIN_;
  int b = e / (NPOS_ * CIN_);
  xh[(b * NPOS_ + n) * CIN_ + c] = f2bf(x[e]);
}

// ------------------------------------------------------------------
// Stage 0b: weights [O][C][27] f32 -> bf16 A-fragments, layout
// [mtile(6)][kblock(54)][lane(32)][half(16)]; K ordering = tap*64 + c.
// 16-bit fragment map: M = lane&15 ; K = 16*(h>>3) + 8*(lane>>4) + (h&7)
// ------------------------------------------------------------------
__global__ __launch_bounds__(256) void k_swizzle_w(const float* __restrict__ wsrc,
                                                   unsigned short* __restrict__ wdst,
                                                   int olim) {
  int e = blockIdx.x * 256 + threadIdx.x;
  if (e >= 6 * KBLK_ * 32 * 16) return;
  int h    = e & 15;
  int lid  = (e >> 4) & 31;
  int rest = e >> 9;
  int kb   = rest % KBLK_;
  int mt   = rest / KBLK_;
  int M = mt * 16 + (lid & 15);
  int k = kb * 32 + ((h >> 3) << 4) + ((lid >> 4) << 3) + (h & 7);
  int tap = k >> 6;        // K = tap*64 + c
  int c   = k & 63;
  float v = 0.0f;
  if (M < olim) v = wsrc[(M * CIN_ + c) * K27_ + tap];
  wdst[e] = f2bf(v);
}

// ------------------------------------------------------------------
// Stage 1: offset conv as fused im2col + WMMA GEMM, 32-position N-tile.
// Phase 1: build two 1728x16 bf16 B sub-tiles in dynamic LDS in fragment
// layout. Phase 2: waves 0..5 each own a 16-row M-tile; each A fragment
// feeds two WMMAs (sub-tile 0 and 1) -> halves A traffic from L2.
// Output: off_raw [B][96][n] f32 (rows 81..95 are the zero-padded rows).
// ------------------------------------------------------------------
__global__ __launch_bounds__(256) void k_offset_gemm(const unsigned short* __restrict__ xh16,
                                                     const unsigned* __restrict__ woff_sw,
                                                     const float* __restrict__ boff,
                                                     float* __restrict__ off_raw) {
  extern __shared__ __align__(32) unsigned ldsB[];
  const unsigned* xh32 = (const unsigned*)xh16;
  int blk  = blockIdx.x;
  int b    = blk / NTILES_;
  int tile = blk % NTILES_;
  int n0   = tile * NT_;
  int tid = threadIdx.x, wave = tid >> 5, lid = tid & 31;

  for (int task = wave; task < NT_ * K27_; task += 8) {   // 864 tasks, 108/wave
    int pos = task & 31;
    int k   = task >> 5;
    int n   = n0 + pos;
    int t   = n / (H_ * W_);
    int rem = n % (H_ * W_);
    int y   = rem / W_;
    int xx  = rem % W_;
    int kt = k / 9, ky = (k / 3) % 3, kx = k % 3;
    int tt = t + kt - 1, yy = y + ky - 1, xw = xx + kx - 1;
    unsigned v = 0u;
    if ((unsigned)tt < (unsigned)T_ && (unsigned)yy < (unsigned)H_ &&
        (unsigned)xw < (unsigned)W_) {
      int idx = (tt * H_ + yy) * W_ + xw;
      v = xh32[(b * NPOS_ + idx) * (CIN_ / 2) + lid];   // channels 2*lid, 2*lid+1
    }
    // scatter channel-pair into B-fragment layout: K = k*64 + 2*lid
    int s     = pos >> 4;                 // sub-tile
    int p16   = pos & 15;
    int kb    = 2 * k + (lid >> 4);
    int lanep = ((lid >> 2) & 1) * 16 + p16;
    int hq    = ((lid >> 3) & 1) * 4 + (lid & 3);       // u32 index (h/2)
    ldsB[s * (KBLK_ * 256) + kb * 256 + lanep * 8 + hq] = v;
  }
  __syncthreads();

  if (wave < 6) {
    v8f acc0 = {}, acc1 = {};
    const v16bf* aptr = (const v16bf*)woff_sw;
    const v16bf* bptr = (const v16bf*)ldsB;
    for (int kb = 0; kb < KBLK_; ++kb) {
      v16bf a  = aptr[(wave * KBLK_ + kb) * 32 + lid];
      v16bf b0 = bptr[kb * 32 + lid];
      v16bf b1 = bptr[(KBLK_ + kb) * 32 + lid];
      acc0 = __builtin_amdgcn_wmma_f32_16x16x32_bf16(false, a, false, b0,
                                                     (short)0, acc0, false, false);
      acc1 = __builtin_amdgcn_wmma_f32_16x16x32_bf16(false, a, false, b1,
                                                     (short)0, acc1, false, false);
    }
    int ng = lid >> 4, ncol = lid & 15;
#pragma unroll
    for (int r = 0; r < 8; ++r) {
      int M = wave * 16 + r + 8 * ng;
      float bias = (M < 81) ? boff[M] : 0.0f;
      float* row = off_raw + (b * COUT_ + M) * NPOS_ + n0 + ncol;
      row[0]  = acc0[r] + bias;
      row[16] = acc1[r] + bias;
    }
  }
}

// ------------------------------------------------------------------
// Stage 2: fused tanh-offset trilinear sampling + deformable-conv GEMM.
// Phase 1 builds the sampled B sub-tiles directly in LDS fragment layout,
// Phase 2 is the identical 6-wave x 54-step dual-WMMA loop with the main
// weights. Output: out_raw [B][96][n] f32 (pre-BN).
// ------------------------------------------------------------------
__global__ __launch_bounds__(256) void k_deform_gemm(const unsigned short* __restrict__ xh16,
                                                     const unsigned* __restrict__ w_sw,
                                                     const float* __restrict__ off_raw,
                                                     float* __restrict__ out_raw) {
  extern __shared__ __align__(32) unsigned ldsB[];
  const unsigned* xh32 = (const unsigned*)xh16;
  int blk  = blockIdx.x;
  int b    = blk / NTILES_;
  int tile = blk % NTILES_;
  int n0   = tile * NT_;
  int tid = threadIdx.x, wave = tid >> 5, lid = tid & 31;

  for (int task = wave; task < NT_ * K27_; task += 8) {
    int pos = task & 31;
    int k   = task >> 5;
    int n   = n0 + pos;
    int t   = n / (H_ * W_);
    int rem = n % (H_ * W_);
    int y   = rem / W_;
    int xx  = rem % W_;
    int kt = k / 9, ky = (k / 3) % 3, kx = k % 3;
    // offsets: channel = comp*27 + k (uniform per wave -> scalar loads)
    float o0 = off_raw[(b * COUT_ + k     ) * NPOS_ + n];
    float o1 = off_raw[(b * COUT_ + 27 + k) * NPOS_ + n];
    float o2 = off_raw[(b * COUT_ + 54 + k) * NPOS_ + n];
    float pt = (float)(t  + kt - 1) + tanhf(o0) * 1.0f;
    float py = (float)(y  + ky - 1) + tanhf(o1) * 2.0f;
    float px = (float)(xx + kx - 1) + tanhf(o2) * 2.0f;
    float tf = floorf(pt), yf = floorf(py), xf = floorf(px);
    float ft = pt - tf, fy = py - yf, fx = px - xf;
    int it = (int)tf, iy = (int)yf, ix = (int)xf;

    float a0 = 0.0f, a1 = 0.0f;   // this lane's two channels
#pragma unroll
    for (int at = 0; at < 2; ++at)
#pragma unroll
      for (int ay = 0; ay < 2; ++ay)
#pragma unroll
        for (int ax = 0; ax < 2; ++ax) {
          int ti = it + at, yi = iy + ay, xi = ix + ax;
          float wgt = (at ? ft : 1.0f - ft) * (ay ? fy : 1.0f - fy) *
                      (ax ? fx : 1.0f - fx);
          bool valid = ((unsigned)ti < (unsigned)T_) &&
                       ((unsigned)yi < (unsigned)H_) &&
                       ((unsigned)xi < (unsigned)W_);
          wgt = valid ? wgt : 0.0f;
          int tc = min(max(ti, 0), T_ - 1);
          int yc = min(max(yi, 0), H_ - 1);
          int xc = min(max(xi, 0), W_ - 1);
          int idx = (tc * H_ + yc) * W_ + xc;
          unsigned v = xh32[(b * NPOS_ + idx) * (CIN_ / 2) + lid];
          a0 += wgt * __uint_as_float(v << 16);
          a1 += wgt * __uint_as_float(v & 0xffff0000u);
        }
    unsigned pk = (unsigned)f2bf(a0) | ((unsigned)f2bf(a1) << 16);
    int s     = pos >> 4;
    int p16   = pos & 15;
    int kb    = 2 * k + (lid >> 4);
    int lanep = ((lid >> 2) & 1) * 16 + p16;
    int hq    = ((lid >> 3) & 1) * 4 + (lid & 3);
    ldsB[s * (KBLK_ * 256) + kb * 256 + lanep * 8 + hq] = pk;
  }
  __syncthreads();

  if (wave < 6) {
    v8f acc0 = {}, acc1 = {};
    const v16bf* aptr = (const v16bf*)w_sw;
    const v16bf* bptr = (const v16bf*)ldsB;
    for (int kb = 0; kb < KBLK_; ++kb) {
      v16bf a  = aptr[(wave * KBLK_ + kb) * 32 + lid];
      v16bf b0 = bptr[kb * 32 + lid];
      v16bf b1 = bptr[(KBLK_ + kb) * 32 + lid];
      acc0 = __builtin_amdgcn_wmma_f32_16x16x32_bf16(false, a, false, b0,
                                                     (short)0, acc0, false, false);
      acc1 = __builtin_amdgcn_wmma_f32_16x16x32_bf16(false, a, false, b1,
                                                     (short)0, acc1, false, false);
    }
    int ng = lid >> 4, ncol = lid & 15;
#pragma unroll
    for (int r = 0; r < 8; ++r) {
      int M = wave * 16 + r + 8 * ng;
      float* row = out_raw + (b * COUT_ + M) * NPOS_ + n0 + ncol;
      row[0]  = acc0[r];
      row[16] = acc1[r];
    }
  }
}

// ------------------------------------------------------------------
// Stage 3: BatchNorm statistics per channel (mean + rsqrt(var+eps)).
// ------------------------------------------------------------------
__global__ __launch_bounds__(256) void k_bn_stats(const float* __restrict__ raw,
                                                  float* __restrict__ mu,
                                                  float* __restrict__ rs) {
  __shared__ float s1[256];
  __shared__ float s2[256];
  int o = blockIdx.x, tid = threadIdx.x;
  float a = 0.0f, q = 0.0f;
  for (int b = 0; b < B_; ++b) {
    const float* p = raw + (b * COUT_ + o) * NPOS_;
    for (int n = tid; n < NPOS_; n += 256) {
      float v = p[n];
      a += v;
      q += v * v;
    }
  }
  s1[tid] = a; s2[tid] = q;
  __syncthreads();
  for (int s = 128; s > 0; s >>= 1) {
    if (tid < s) { s1[tid] += s1[tid + s]; s2[tid] += s2[tid + s]; }
    __syncthreads();
  }
  if (tid == 0) {
    float inv = 1.0f / (float)(B_ * NPOS_);
    float m   = s1[0] * inv;
    float var = s2[0] * inv - m * m;
    mu[o] = m;
    rs[o] = rsqrtf(var + 1e-5f);
  }
}

// ------------------------------------------------------------------
// Stage 4: BN apply + ReLU -> d_out [B][COUT][T][H][W] (same flat order).
// ------------------------------------------------------------------
__global__ __launch_bounds__(256) void k_bn_apply(const float* __restrict__ raw,
                                                  const float* __restrict__ mu,
                                                  const float* __restrict__ rs,
                                                  const float* __restrict__ gamma,
                                                  const float* __restrict__ beta,
                                                  float* __restrict__ out) {
  int e = blockIdx.x * 256 + threadIdx.x;
  if (e >= B_ * COUT_ * NPOS_) return;
  int o = (e / NPOS_) % COUT_;
  float v = (raw[e] - mu[o]) * rs[o] * gamma[o] + beta[o];
  out[e] = fmaxf(v, 0.0f);
}

// ------------------------------------------------------------------
extern "C" void kernel_launch(void* const* d_in, const int* in_sizes, int n_in,
                              void* d_out, int out_size, void* d_ws, size_t ws_size,
                              hipStream_t stream) {
  const float* x     = (const float*)d_in[0];
  const float* w_off = (const float*)d_in[1];
  const float* b_off = (const float*)d_in[2];
  const float* w     = (const float*)d_in[3];
  const float* gamma = (const float*)d_in[4];
  const float* beta  = (const float*)d_in[5];

  // workspace layout (128B-aligned regions)
  char* ws = (char*)d_ws;
  unsigned short* xh      = (unsigned short*)(ws + 0);          //  4,718,592 B
  unsigned short* woff_sw = (unsigned short*)(ws + 4718592);    //    331,776 B
  unsigned short* w_sw    = (unsigned short*)(ws + 5050368);    //    331,776 B
  float* off_raw          = (float*)(ws + 5382144);             // 14,155,776 B
  float* out_raw          = (float*)(ws + 19537920);            // 14,155,776 B
  float* mu               = (float*)(ws + 33693696);            //        384 B
  float* rs               = (float*)(ws + 33694080);            //        384 B

  k_convert_x<<<(B_ * CIN_ * NPOS_ + 255) / 256, 256, 0, stream>>>(x, xh);
  k_swizzle_w<<<(6 * KBLK_ * 512 + 255) / 256, 256, 0, stream>>>(w_off, woff_sw, 81);
  k_swizzle_w<<<(6 * KBLK_ * 512 + 255) / 256, 256, 0, stream>>>(w, w_sw, 96);
  k_offset_gemm<<<B_ * NTILES_, 256, LDS_BYTES_, stream>>>(
      xh, (const unsigned*)woff_sw, b_off, off_raw);
  k_deform_gemm<<<B_ * NTILES_, 256, LDS_BYTES_, stream>>>(
      xh, (const unsigned*)w_sw, off_raw, out_raw);
  k_bn_stats<<<COUT_, 256, 0, stream>>>(out_raw, mu, rs);
  k_bn_apply<<<(B_ * COUT_ * NPOS_ + 255) / 256, 256, 0, stream>>>(
      out_raw, mu, rs, gamma, beta, (float*)d_out);
}
